// ScannedRNN_10282151706818
// MI455X (gfx1250) — compile-verified, matrix operand order checked by reading, not run on hardware
//
#include <hip/hip_runtime.h>

typedef __bf16 v16bf __attribute__((ext_vector_type(16)));
typedef float  v8f   __attribute__((ext_vector_type(8)));
typedef unsigned short u16;

#define T_STEPS 1024
#define BATCH   256
#define HID     256
#define NCOLS   1024            // 4*HID
#define KT_N    8               // K=256 -> 8 k-tiles of 32 (both GEMMs)
#define NT_N    64              // 1024 cols -> 64 n-tiles of 16

// -------- workspace layout --------
// [0, 512KB)       : packed Wi  bf16  (8 kt x 64 nt x 32 lane x 16 e)
// [512KB, 1MB)     : packed Wh  bf16
// [1MB, 1MB+1GB)   : Zx = xs@Wi + b, fp32, WMMA C/D fragment layout:
//                    [rowtile (16384)][ntile (64)][lane (32)][v (8)]
#define WPK_ELEMS (KT_N * NT_N * 32 * 16)        // 262144 per matrix
#define ZX_OFF_BYTES (2u * WPK_ELEMS * 2u)       // 1 MB

// -------- scan-kernel LDS carve-up --------
#define AFRAG_BYTES (KT_N * 32 * 16 * 2)         // 8 KB: h fragments [kt][lane][e]
#define Z_STRIDE    1028                         // padded vs 64 banks
#define Z_OFF       AFRAG_BYTES
#define Z_BYTES     (16 * Z_STRIDE * 4)          // 65792 B
#define C_OFF       (Z_OFF + Z_BYTES)
#define C_BYTES     (16 * HID * 4)
#define SMEM_BYTES  (C_OFF + C_BYTES)            // 90368 B (< 320 KB WGP LDS)

__device__ __forceinline__ u16 f2bf(float f) {
    unsigned int u = __float_as_uint(f);
    unsigned int r = u + 0x7FFFu + ((u >> 16) & 1u);   // RNE
    return (u16)(r >> 16);
}

// 16-bit 16x32 A-matrix lane/element position (ISA 7.12.2)
__device__ __forceinline__ void frag_pos(int m, int kp, int& ln, int& e) {
    int grp, v, half = kp & 1;
    if (kp < 8)       { grp = 0; v = kp >> 1; }
    else if (kp < 16) { grp = 1; v = (kp - 8) >> 1; }
    else if (kp < 24) { grp = 0; v = 4 + ((kp - 16) >> 1); }
    else              { grp = 1; v = 4 + ((kp - 24) >> 1); }
    ln = grp * 16 + m;
    e  = 2 * v + half;
}

__device__ __forceinline__ float sigm(float x) { return 1.0f / (1.0f + __expf(-x)); }
__device__ __forceinline__ float tanh_f(float x) { return 2.0f * sigm(2.0f * x) - 1.0f; }

// ---- Kernel 0: pack Wi and Wh (256x1024 fp32) into bf16 B-fragment order ----
__global__ void lstm_pack_weights(const float* __restrict__ Wi,
                                  const float* __restrict__ Wh,
                                  u16* __restrict__ Wpk) {
    int idx = blockIdx.x * 256 + threadIdx.x;          // 0 .. 524287 (both matrices)
    int e     = idx & 15;
    int lane  = (idx >> 4) & 31;
    int ntile = (idx >> 9) & 63;
    int kt    = (idx >> 15) & 7;
    int sel   = idx >> 18;                             // 0 = Wi, 1 = Wh
    int n     = ntile * 16 + (lane & 15);
    int grp   = lane >> 4;
    int v     = e >> 1, half = e & 1;
    int kp    = (v < 4) ? (2 * v + half + grp * 8)
                        : (16 + 2 * (v - 4) + half + grp * 8);
    size_t k  = (size_t)kt * 32 + kp;                  // 0..255
    const float* W = sel ? Wh : Wi;
    Wpk[idx] = f2bf(W[k * NCOLS + n]);
}

// ---- Kernel 1: Zx = xs @ Wi + b, full-chip parallel, fragment-layout output ----
// grid = 8192 WGs, each handles TWO 16-row tiles (halves L2 weight traffic),
// 512 threads = 16 waves, wave w -> ntiles w*4 .. w*4+3, double-buffered B.
__global__ void lstm_zx_gemm(const float* __restrict__ xs,
                             const u16* __restrict__ Wpk_i,
                             const float* __restrict__ bias,
                             float* __restrict__ Zx) {
    __shared__ u16 afrag[2 * KT_N * 32 * 16];          // two A tiles, 16 KB
    const int tid  = threadIdx.x;                      // 0..511
    const int lane = tid & 31;
    const int wave = tid >> 5;                         // 0..15
    const size_t rt0 = (size_t)blockIdx.x * 2;         // first of two row tiles

    // stage 32 rows of x (2 tiles) as bf16 A fragments
    {
        int m  = tid >> 4;                             // 0..31
        int k0 = (tid & 15) << 4;                      // 0..240
        int rr = m >> 4;                               // which tile
        int ml = m & 15;                               // row within tile
        const float4* xp = (const float4*)(xs + ((rt0 + rr) * 16 + ml) * HID + k0);
        float4 q[4];
        q[0] = xp[0]; q[1] = xp[1]; q[2] = xp[2]; q[3] = xp[3];
        const float* qs = (const float*)q;
        u16* af = afrag + rr * (KT_N * 32 * 16);
        #pragma unroll
        for (int p = 0; p < 8; ++p) {
            int k = k0 + 2 * p;
            int kt = k >> 5, kp = k & 31, ln, e;
            frag_pos(ml, kp, ln, e);
            u16* dst = &af[(kt * 32 + ln) * 16 + e];
            dst[0] = f2bf(qs[2 * p]);
            dst[1] = f2bf(qs[2 * p + 1]);
        }
    }
    __syncthreads();

    v8f acc0[4], acc1[4];
    #pragma unroll
    for (int nt = 0; nt < 4; ++nt) { acc0[nt] = 0.0f; acc1[nt] = 0.0f; }

    const v16bf* Bb = (const v16bf*)Wpk_i;
    const size_t lbase = (size_t)(wave * 4) * 32 + lane;

    v16bf bcur[4], bnxt[4];
    #pragma unroll
    for (int nt = 0; nt < 4; ++nt) bcur[nt] = Bb[lbase + nt * 32];
    #pragma unroll
    for (int kt = 0; kt < KT_N; ++kt) {
        if (kt < KT_N - 1) {
            #pragma unroll
            for (int nt = 0; nt < 4; ++nt)
                bnxt[nt] = Bb[lbase + (size_t)(kt + 1) * (NT_N * 32) + nt * 32];
        }
        v16bf a0 = *(const v16bf*)&afrag[(kt * 32 + lane) * 16];
        v16bf a1 = *(const v16bf*)&afrag[(KT_N * 32 * 16) + (kt * 32 + lane) * 16];
        #pragma unroll
        for (int nt = 0; nt < 4; ++nt) {
            acc0[nt] = __builtin_amdgcn_wmma_f32_16x16x32_bf16(
                false, a0, false, bcur[nt], (short)0, acc0[nt], false, false);
            acc1[nt] = __builtin_amdgcn_wmma_f32_16x16x32_bf16(
                false, a1, false, bcur[nt], (short)0, acc1[nt], false, false);
        }
        if (kt < KT_N - 1) {
            #pragma unroll
            for (int nt = 0; nt < 4; ++nt) bcur[nt] = bnxt[nt];
        }
    }

    // fold bias, store both tiles in C/D fragment layout: [rowtile][ntile][lane][v]
    #pragma unroll
    for (int nt = 0; nt < 4; ++nt) {
        int ntile = wave * 4 + nt;
        float bv = bias[ntile * 16 + (lane & 15)];
        v8f r0 = acc0[nt], r1 = acc1[nt];
        #pragma unroll
        for (int v = 0; v < 8; ++v) { r0[v] += bv; r1[v] += bv; }
        *(v8f*)(Zx + (((rt0 + 0) * 64 + ntile) * 32 + lane) * 8) = r0;
        *(v8f*)(Zx + (((rt0 + 1) * 64 + ntile) * 32 + lane) * 8) = r1;
    }
}

// ---- Kernel 2: persistent recurrent scan, z = Zx + h@Wh ----
// grid = 16 WGs (16-row batch slices), 512 threads = 16 waves (4 waves/SIMD
// for L2-latency hiding); wave w owns ntiles w*4 .. w*4+3 (64 gate columns).
__global__ void lstm_scan(const unsigned char* __restrict__ resets,
                          const float* __restrict__ c0,
                          const float* __restrict__ h0,
                          const u16* __restrict__ Wpk_h,
                          const float* __restrict__ Zx,
                          float* __restrict__ ys) {
    extern __shared__ char smem[];
    u16*   afrag = (u16*)(smem);                 // [8 kt][32 lane][16 e] h fragments
    float* zbuf  = (float*)(smem + Z_OFF);       // [16 m][1028]
    float* cbuf  = (float*)(smem + C_OFF);       // [16 m][256]

    const int tid   = threadIdx.x;               // 0..511
    const int lane  = tid & 31;
    const int wave  = tid >> 5;                  // 0..15
    const int bid   = blockIdx.x;
    const int brow0 = bid * 16;

    // one-time init: c-state and h0 fragments
    for (int i = tid; i < 16 * HID; i += 512) cbuf[i] = c0[(size_t)brow0 * HID + i];
    for (int i = tid; i < 16 * HID; i += 512) {
        int m = i >> 8, j = i & 255;
        int kt = j >> 5, kp = j & 31, ln, e;
        frag_pos(m, kp, ln, e);
        afrag[(kt * 32 + ln) * 16 + e] = f2bf(h0[(size_t)(brow0 + m) * HID + j]);
    }
    __syncthreads();

    const v16bf* Bb = (const v16bf*)Wpk_h;
    const size_t lbase = (size_t)(wave * 4) * 32 + lane;

    for (int t = 0; t < T_STEPS; ++t) {
        // ---- Phase A: reference masks h BEFORE the GEMM -> zero reset rows ----
        if (tid < 256) {
            int m2 = tid & 15, s = tid >> 4;     // 16 slices cover 8 kt x 2 lane halves
            if (resets[(size_t)t * BATCH + brow0 + m2]) {
                int kt = s >> 1;
                int ln = ((s & 1) << 4) + m2;
                uint4* p0 = (uint4*)&afrag[(kt * 32 + ln) * 16];
                p0[0] = make_uint4(0, 0, 0, 0);
                p0[1] = make_uint4(0, 0, 0, 0);
            }
        }
        __syncthreads();

        // ---- Phase B: acc = Zx(t) fragments, then += h @ Wh (pipelined B stream) ----
        v8f acc[4];
        const float* zx = Zx + ((((size_t)t * 16 + bid) * 64 + wave * 4) * 32 + lane) * 8;
        #pragma unroll
        for (int nt = 0; nt < 4; ++nt)
            acc[nt] = *(const v8f*)(zx + nt * 32 * 8);

        v16bf bcur[4], bnxt[4];
        #pragma unroll
        for (int nt = 0; nt < 4; ++nt) bcur[nt] = Bb[lbase + nt * 32];
        #pragma unroll
        for (int kt = 0; kt < KT_N; ++kt) {
            if (kt < KT_N - 1) {
                #pragma unroll
                for (int nt = 0; nt < 4; ++nt)
                    bnxt[nt] = Bb[lbase + (size_t)(kt + 1) * (NT_N * 32) + nt * 32];
            }
            v16bf a = *(const v16bf*)&afrag[(kt * 32 + lane) * 16];
            #pragma unroll
            for (int nt = 0; nt < 4; ++nt)
                acc[nt] = __builtin_amdgcn_wmma_f32_16x16x32_bf16(
                    false, a, false, bcur[nt], (short)0, acc[nt], false, false);
            if (kt < KT_N - 1) {
                #pragma unroll
                for (int nt = 0; nt < 4; ++nt) bcur[nt] = bnxt[nt];
            }
        }

        // ---- Phase C: spill z tiles to LDS (C/D layout) ----
        {
            int r    = (lane < 16) ? 0 : 8;
            int nloc = lane & 15;
            #pragma unroll
            for (int nt = 0; nt < 4; ++nt) {
                int nbase = (wave * 4 + nt) * 16 + nloc;
                #pragma unroll
                for (int v = 0; v < 8; ++v)
                    zbuf[(size_t)(v + r) * Z_STRIDE + nbase] = acc[nt][v];
            }
        }
        __syncthreads();

        // ---- Phase D: gates, state update, h_new -> ys + next fragments ----
        {
            int j  = tid & 255;                  // hidden column
            int m0 = (tid >> 8) * 8;             // rows 0..7 or 8..15
            // pull next step's Zx slice (64 KB) toward L2 while doing scalar math
            if (t + 1 < T_STEPS)
                __builtin_prefetch(Zx + (((size_t)(t + 1) * 16 + bid) * 16384) + tid * 32, 0, 1);
            #pragma unroll 4
            for (int mm = 0; mm < 8; ++mm) {
                int m = m0 + mm;
                float zi = zbuf[(size_t)m * Z_STRIDE + j];
                float zf = zbuf[(size_t)m * Z_STRIDE + HID + j];
                float zg = zbuf[(size_t)m * Z_STRIDE + 2 * HID + j];
                float zo = zbuf[(size_t)m * Z_STRIDE + 3 * HID + j];
                float cold = resets[(size_t)t * BATCH + brow0 + m] ? 0.0f
                                                                   : cbuf[m * HID + j];
                float cn = sigm(zf) * cold + sigm(zi) * tanh_f(zg);
                float hn = sigm(zo) * tanh_f(cn);
                cbuf[m * HID + j] = cn;
                ys[((size_t)t * BATCH + brow0 + m) * HID + j] = hn;
                int kt = j >> 5, kp = j & 31, ln, e;
                frag_pos(m, kp, ln, e);
                afrag[(kt * 32 + ln) * 16 + e] = f2bf(hn);
            }
        }
        __syncthreads();
    }
}

extern "C" void kernel_launch(void* const* d_in, const int* in_sizes, int n_in,
                              void* d_out, int out_size, void* d_ws, size_t ws_size,
                              hipStream_t stream) {
    const float*         xs     = (const float*)d_in[0];
    const unsigned char* resets = (const unsigned char*)d_in[1];  // jax bool = 1 byte
    const float*         c0     = (const float*)d_in[2];
    const float*         h0     = (const float*)d_in[3];
    const float*         Wi     = (const float*)d_in[4];
    const float*         Wh     = (const float*)d_in[5];
    const float*         b      = (const float*)d_in[6];
    float*               ys     = (float*)d_out;

    u16*   Wpk_i = (u16*)d_ws;
    u16*   Wpk_h = Wpk_i + WPK_ELEMS;
    float* Zx    = (float*)((char*)d_ws + ZX_OFF_BYTES);          // 1 GB fp32

    lstm_pack_weights<<<2048, 256, 0, stream>>>(Wi, Wh, Wpk_i);
    lstm_zx_gemm<<<8192, 512, 0, stream>>>(xs, Wpk_i, b, Zx);     // full-chip parallel
    lstm_scan<<<16, 512, SMEM_BYTES, stream>>>(resets, c0, h0, Wpk_h, Zx, ys);
}
